// GNN_26276609917064
// MI455X (gfx1250) — matrix operand order repaired and checked by reading, not compiled, not run on hardware
//
#include <hip/hip_runtime.h>
#include <hip/hip_bf16.h>

// ---------------------------------------------------------------------------
// GNN (3x GCN + TopK pooling + readout) + MLP head, tuned for MI455X/gfx1250.
//  - 1 workgroup (256 thr = 8 wave32) per graph; all graph state in LDS
//    (~211 KB of the 320 KB CDNA5 WGP budget -> whole pipeline on-chip).
//  - x / edge staging via GLOBAL_LOAD_ASYNC_TO_LDS (ASYNCcnt) when available.
//  - GEMMs x@W via v_wmma_f32_16x16x32_f16; A and B both pre-staged as f16 in
//    LDS with bank-skewed strides -> conflict-free ds_load_b128 fragments.
//  - Scatter-aggregate via ds_add_f32 LDS atomics; TopK via LDS bitonic sort.
// ---------------------------------------------------------------------------

typedef _Float16 h16;
typedef __attribute__((ext_vector_type(16))) _Float16 v16h;
typedef __attribute__((ext_vector_type(8)))  float    v8f;
typedef __attribute__((ext_vector_type(4)))  int      v4i_t;

#define NB    256          // max nodes per graph
#define NE    2048         // edges per graph
#define NEMB  39           // input embed dim
#define HF    64           // hidden dim
#define SX    68           // f32 row stride (68*4B -> +4 banks/row skew)
#define SXH   72           // f16 row stride for staged A (144B -> +36 banks/row)
#define WS    72           // f16 row stride for W^T     (144B -> +36 banks/row)

#if defined(__gfx1250__) && \
    __has_builtin(__builtin_amdgcn_global_load_async_to_lds_b32) && \
    __has_builtin(__builtin_amdgcn_global_load_async_to_lds_b128) && \
    __has_builtin(__builtin_amdgcn_s_wait_asynccnt)
#define USE_ASYNC_LDS 1
typedef __attribute__((address_space(1))) int   g_i32;     // global int
typedef __attribute__((address_space(3))) int   l_i32;     // LDS int
typedef __attribute__((address_space(1))) v4i_t g_v4i;     // global int4
typedef __attribute__((address_space(3))) v4i_t l_v4i;     // LDS int4
#endif

// ---- WMMA GEMM: bh[0:rowTiles*16, 0:64] = A(f16) @ B(f16), f32 acc ---------
// A frag (16x32 f16): lane m=L&15; halves j<8 -> k=kb+j, j>=8 -> k=kb+16+(j-8),
//                     kb = (L>=16 ? 8 : 0)   (ISA 7.12.2, 16-bit A layout)
// B frag (32x16 f16): lane n=L&15; same k order as A (kb = L>=16 ? 8 : 0)
// C/D   (16x16 f32):  VGPR i, lane L: n=L&15, m = i + (L>=16 ? 8 : 0)
__device__ __forceinline__ void gemm_wmma(const h16* __restrict__ xh,
                                          float* __restrict__ bh,
                                          const h16* __restrict__ wt,
                                          int rowTiles, int lane, int wave)
{
    const int nTiles = rowTiles * 4;            // 4 col tiles (64 outputs)
    for (int t = wave; t < nTiles; t += 8) {    // wave-uniform: EXEC stays all-1
        const int tm = t >> 2, tn = t & 3;
        const int m    = tm * 16 + (lane & 15);
        const int ncol = tn * 16 + (lane & 15);
        const int kbA  = (lane >> 4) << 3;      // 0 or 8
        const int kbB  = (lane >> 4) << 3;      // 0 or 8
        v8f acc = {};
#pragma unroll
        for (int kk = 0; kk < 64; kk += 32) {
            const h16* pa = &xh[m * SXH + kk + kbA];     // 2x 16B ds_load_b128
            v16h a;
#pragma unroll
            for (int j = 0; j < 8; ++j) { a[j] = pa[j]; a[8 + j] = pa[16 + j]; }
            const h16* pw = &wt[ncol * WS + kk + kbB];   // 2x 16B ds_load_b128
            v16h b;
#pragma unroll
            for (int j = 0; j < 8; ++j) { b[j] = pw[j]; b[8 + j] = pw[16 + j]; }

            acc = __builtin_amdgcn_wmma_f32_16x16x32_f16(
                false, a, false, b, (short)0, acc, false, false);
        }
        const int nloc  = tn * 16 + (lane & 15);
        const int mbase = tm * 16 + ((lane >> 4) << 3);
#pragma unroll
        for (int i = 0; i < 8; ++i)
            bh[(mbase + i) * SX + nloc] = acc[i];
    }
}

// ---- bitonic sort of 256 (key,idx) pairs: key desc, ties -> idx asc --------
__device__ __forceinline__ void bitonic_desc(float* key, int* idx, int tid)
{
    for (int ksz = 2; ksz <= NB; ksz <<= 1) {
        for (int j = ksz >> 1; j > 0; j >>= 1) {
            __syncthreads();
            int l = tid ^ j;
            if (l > tid) {
                float a = key[tid], b = key[l];
                int   ia = idx[tid], ib = idx[l];
                bool a_first = (a > b) || (a == b && ia < ib);
                bool up = ((tid & ksz) == 0);      // descending sub-sequence
                if (up ? !a_first : a_first) {
                    key[tid] = b; key[l] = a;
                    idx[tid] = ib; idx[l] = ia;
                }
            }
        }
    }
    __syncthreads();
}

__global__ __launch_bounds__(256) void gnn_graph_kernel(
    const float* __restrict__ x, const int* __restrict__ ei,
    const float* __restrict__ W1, const float* __restrict__ b1, const float* __restrict__ p1,
    const float* __restrict__ W2, const float* __restrict__ b2, const float* __restrict__ p2,
    const float* __restrict__ W3, const float* __restrict__ b3, const float* __restrict__ p3,
    float* __restrict__ feats)
{
    __shared__ __attribute__((aligned(16))) float buf0[NB * SX];
    __shared__ __attribute__((aligned(16))) float buf1[NB * SX];
    __shared__ __attribute__((aligned(16))) h16   xh[NB * SXH];   // f16 A stage
    __shared__ __attribute__((aligned(16))) h16   wt[HF * WS];    // f16 W^T
    __shared__ float deg[NB], dinv[NB];
    __shared__ float skey[NB];
    __shared__ int   sidx[NB], nid[NB];
    __shared__ __attribute__((aligned(16))) int   esrc[NE], edst[NE];
    __shared__ float emk[NE];
    __shared__ float racc[128];
    __shared__ float pl[HF];
    __shared__ float pns;

    const int tid  = threadIdx.x;
    const int lane = tid & 31;
    const int wave = tid >> 5;
    const int g    = blockIdx.x;

    const float* xg = x + (size_t)g * NB * NEMB;
    const int*   eg = ei + (size_t)g * 2 * NE;

#ifdef USE_ASYNC_LDS
    // x: 256 rows x 39 f32, direct HBM -> LDS (ASYNCcnt path, no VGPR bounce)
    for (int t = tid; t < NB * NEMB; t += 256) {
        int i = t / NEMB, f = t - i * NEMB;
        __builtin_amdgcn_global_load_async_to_lds_b32(
            (g_i32*)(xg + i * NEMB + f),
            (l_i32*)(&buf0[i * SX + f]), 0, 0);
    }
    // edges: 2x 8KB contiguous, 16B-aligned -> b128 async copies
    for (int t = tid; t < NE / 4; t += 256)
        __builtin_amdgcn_global_load_async_to_lds_b128(
            (g_v4i*)(eg + t * 4), (l_v4i*)(&esrc[t * 4]), 0, 0);
    for (int t = tid; t < NE / 4; t += 256)
        __builtin_amdgcn_global_load_async_to_lds_b128(
            (g_v4i*)(eg + NE + t * 4), (l_v4i*)(&edst[t * 4]), 0, 0);
    // zero-pad cols 39..63 (disjoint from async writes) while DMA is in flight
    for (int t = tid; t < NB * (HF - NEMB); t += 256) {
        int i = t / (HF - NEMB), f = NEMB + (t - i * (HF - NEMB));
        buf0[i * SX + f] = 0.f;
    }
    for (int e = tid; e < NE; e += 256) emk[e] = 1.f;
    __builtin_amdgcn_s_wait_asynccnt(0);
#else
    for (int jj = tid; jj < NB * HF; jj += 256) {
        int i = jj >> 6, f = jj & 63;
        buf0[i * SX + f] = (f < NEMB) ? xg[i * NEMB + f] : 0.f;
    }
    for (int e = tid; e < NE; e += 256) {
        esrc[e] = eg[e];
        edst[e] = eg[NE + e];
        emk[e]  = 1.f;
    }
#endif
    for (int f = tid; f < 128; f += 256) racc[f] = 0.f;
    __syncthreads();

    const float* Ws[3] = {W1, W2, W3};
    const float* bs[3] = {b1, b2, b3};
    const float* ps[3] = {p1, p2, p3};
    const int    ks[3] = {205, 164, 132};   // ceil(0.8*{256,205,164})

    int n = NB;
#pragma unroll
    for (int layer = 0; layer < 3; ++layer) {
        float* bx = (layer & 1) ? buf1 : buf0;   // current features
        float* bh = (layer & 1) ? buf0 : buf1;   // h / pooled scratch
        const int Kdim = (layer == 0) ? NEMB : HF;
        const int k    = ks[layer];
        const int rowTiles = (n + 15) >> 4;

        // prefetch next layer's params into L2 while we work
        if (layer < 2 && tid < 64) {
            __builtin_prefetch(&Ws[layer + 1][tid * 64], 0, 1);
            if (tid == 0) { __builtin_prefetch(bs[layer + 1], 0, 1);
                            __builtin_prefetch(ps[layer + 1], 0, 1); }
        }
        // stage W^T as f16 into LDS (zero-pad K)
        for (int jj = tid; jj < HF * HF; jj += 256) {
            int col = jj & 63, kr = jj >> 6;
            wt[col * WS + kr] = (kr < Kdim) ? (h16)Ws[layer][kr * HF + col] : (h16)0;
        }
        // degree = 1 + scatter(emask)
        for (int i = tid; i < NB; i += 256) deg[i] = 1.f;
        __syncthreads();
        for (int e = tid; e < NE; e += 256) {
            float m = emk[e];
            if (m > 0.f) atomicAdd(&deg[edst[e]], m);
        }
        __syncthreads();
        for (int i = tid; i < NB; i += 256) dinv[i] = rsqrtf(deg[i]);
        // stage A operand as f16 once (instead of 4x cvt inside the GEMM)
        for (int jj = tid; jj < rowTiles * 16 * HF; jj += 256) {
            int i = jj >> 6, f = jj & 63;
            xh[i * SXH + f] = (h16)bx[i * SX + f];
        }
        __syncthreads();

        // h = x @ W   (WMMA)
        gemm_wmma(xh, bh, wt, rowTiles, lane, wave);
        __syncthreads();

        // zero agg (reuse bx; old x is dead now)
        for (int jj = tid; jj < n * HF; jj += 256) {
            int i = jj >> 6, f = jj & 63;
            bx[i * SX + f] = 0.f;
        }
        __syncthreads();

        // agg[dst] += h[src] * coef   (edge x 16-feature chunks, ds_add_f32)
        for (int t = tid; t < NE * 4; t += 256) {
            int e = t >> 2, c = (t & 3) << 4;
            float m = emk[e];
            if (m > 0.f) {
                int s = esrc[e], d = edst[e];
                float coef = dinv[s] * dinv[d] * m;
#pragma unroll
                for (int f = 0; f < 16; ++f)
                    atomicAdd(&bx[d * SX + c + f], bh[s * SX + c + f] * coef);
            }
        }
        __syncthreads();

        // x = relu(agg + h/deg + b)
        for (int jj = tid; jj < n * HF; jj += 256) {
            int i = jj >> 6, f = jj & 63;
            float v = bx[i * SX + f] + bh[i * SX + f] / deg[i] + bs[layer][f];
            bx[i * SX + f] = v > 0.f ? v : 0.f;
        }

        // ---- TopK pooling ----
        if (tid < HF) pl[tid] = ps[layer][tid];
        __syncthreads();
        if (tid == 0) {
            float s = 0.f;
            for (int f = 0; f < HF; ++f) s += pl[f] * pl[f];
            pns = sqrtf(s);
        }
        __syncthreads();
        for (int i = tid; i < NB; i += 256) {
            if (i < n) {
                float d = 0.f;
                for (int f = 0; f < HF; ++f) d += bx[i * SX + f] * pl[f];
                skey[i] = tanhf(d / pns);
            } else {
                skey[i] = -__builtin_inff();
            }
            sidx[i] = i;
        }
        bitonic_desc(skey, sidx, tid);

        for (int i = tid; i < NB; i += 256) nid[i] = -1;
        __syncthreads();
        for (int i = tid; i < k; i += 256) nid[sidx[i]] = i;
        // gather pooled rows scaled by score -> bh
        for (int jj = tid; jj < k * HF; jj += 256) {
            int i = jj >> 6, f = jj & 63;
            bh[i * SX + f] = bx[sidx[i] * SX + f] * skey[i];
        }
        __syncthreads();
        // edge remap
        for (int e = tid; e < NE; e += 256) {
            int s = esrc[e], d = edst[e];
            int ns = nid[s], nd = nid[d];
            bool valid = (emk[e] > 0.f) && (ns >= 0) && (nd >= 0);
            esrc[e] = ns > 0 ? ns : 0;
            edst[e] = nd > 0 ? nd : 0;
            emk[e]  = valid ? 1.f : 0.f;
        }
        __syncthreads();
        // readout: racc += [max ; mean]
        if (tid < 128) {
            int f = tid & 63;
            if (tid < HF) {
                float mx = -__builtin_inff();
                for (int i = 0; i < k; ++i) { float v = bh[i * SX + f]; mx = v > mx ? v : mx; }
                racc[f] += mx;
            } else {
                float sm = 0.f;
                for (int i = 0; i < k; ++i) sm += bh[i * SX + f];
                racc[64 + f] += sm / (float)k;
            }
        }
        __syncthreads();
        n = k;   // pooled features now live in bh == next layer's bx
    }

    if (tid < 128) feats[(size_t)g * 128 + tid] = racc[tid];
}

// ---- MLP head: sigmoid(relu(relu(f@lW1+lb1)@lW2+lb2)@lW3+lb3) --------------
__global__ __launch_bounds__(256) void mlp_head_kernel(
    const float* __restrict__ feats,
    const float* __restrict__ lW1, const float* __restrict__ lb1,
    const float* __restrict__ lW2, const float* __restrict__ lb2,
    const float* __restrict__ lW3, const float* __restrict__ lb3,
    float* __restrict__ out, int rows)
{
    __shared__ float w1[128 * 64];
    __shared__ float w2[64 * 32];
    __shared__ float w3[32];
    __shared__ float bb1[64], bb2[32], bb3s;
    int tid = threadIdx.x;
    for (int j = tid; j < 128 * 64; j += 256) w1[j] = lW1[j];
    for (int j = tid; j < 64 * 32;  j += 256) w2[j] = lW2[j];
    if (tid < 32) w3[tid] = lW3[tid];
    if (tid < 64) bb1[tid] = lb1[tid];
    if (tid < 32) bb2[tid] = lb2[tid];
    if (tid == 0) bb3s = lb3[0];
    __syncthreads();

    int row = blockIdx.x * 256 + tid;
    if (row >= rows) return;

    float h1[64];
#pragma unroll
    for (int o = 0; o < 64; ++o) h1[o] = bb1[o];
    for (int j = 0; j < 128; ++j) {
        float fv = feats[(size_t)row * 128 + j];
#pragma unroll
        for (int o = 0; o < 64; ++o) h1[o] += fv * w1[j * 64 + o];
    }
#pragma unroll
    for (int o = 0; o < 64; ++o) h1[o] = h1[o] > 0.f ? h1[o] : 0.f;

    float h2[32];
#pragma unroll
    for (int o = 0; o < 32; ++o) h2[o] = bb2[o];
    for (int j = 0; j < 64; ++j) {
        float hv = h1[j];
#pragma unroll
        for (int o = 0; o < 32; ++o) h2[o] += hv * w2[j * 32 + o];
    }
    float s = bb3s;
#pragma unroll
    for (int o = 0; o < 32; ++o) {
        float hv = h2[o] > 0.f ? h2[o] : 0.f;
        s += hv * w3[o];
    }
    out[row] = 1.f / (1.f + __expf(-s));
}

extern "C" void kernel_launch(void* const* d_in, const int* in_sizes, int n_in,
                              void* d_out, int out_size, void* d_ws, size_t ws_size,
                              hipStream_t stream)
{
    const float* x   = (const float*)d_in[0];
    const int*   ei  = (const int*)d_in[1];
    const float* W1  = (const float*)d_in[2];
    const float* b1  = (const float*)d_in[3];
    const float* p1  = (const float*)d_in[4];
    const float* W2  = (const float*)d_in[5];
    const float* b2  = (const float*)d_in[6];
    const float* p2  = (const float*)d_in[7];
    const float* W3  = (const float*)d_in[8];
    const float* b3  = (const float*)d_in[9];
    const float* p3  = (const float*)d_in[10];
    const float* lW1 = (const float*)d_in[11];
    const float* lb1 = (const float*)d_in[12];
    const float* lW2 = (const float*)d_in[13];
    const float* lb2 = (const float*)d_in[14];
    const float* lW3 = (const float*)d_in[15];
    const float* lb3 = (const float*)d_in[16];

    const int B = 1024;
    float* feats = (float*)d_ws;   // B x 128 f32 graph features

    gnn_graph_kernel<<<B, 256, 0, stream>>>(x, ei, W1, b1, p1, W2, b2, p2,
                                            W3, b3, p3, feats);
    mlp_head_kernel<<<(B + 255) / 256, 256, 0, stream>>>(
        feats, lW1, lb1, lW2, lb2, lW3, lb3, (float*)d_out, B);
}
